// LinkPredictor_43276090474888
// MI455X (gfx1250) — matrix operand order repaired and checked by reading, not compiled, not run on hardware
//
#include <hip/hip_runtime.h>
#include <hip/hip_bf16.h>

typedef __attribute__((ext_vector_type(16))) __bf16 bf16x16;
typedef __attribute__((ext_vector_type(8)))  __bf16 bf16x8;
typedef __attribute__((ext_vector_type(8)))  float  f32x8;
typedef __attribute__((ext_vector_type(4)))  float  f32x4;

#define IN_DIM   128
#define KDIM     256          // 2*IN_DIM
#define HIDDEN   128
#define TILE_M   16
#define WAVES_PB 8
#define BLOCK_T  256
#define W1_LD    272          // padded row stride (bf16 elems), 16B-aligned
#define PAIR_LD  272

static __device__ __forceinline__ bf16x16 cat16(bf16x8 lo, bf16x8 hi) {
    return __builtin_shufflevector(lo, hi, 0,1,2,3,4,5,6,7,8,9,10,11,12,13,14,15);
}

__global__ __launch_bounds__(BLOCK_T, 1)
void link_pred_wmma_kernel(const float* __restrict__ z,
                           const long long* __restrict__ edge_src,
                           const long long* __restrict__ edge_dst,
                           const float* __restrict__ W1,   // [128][256] row-major (out,in)
                           const float* __restrict__ b1,   // [128]
                           const float* __restrict__ W2,   // [128]
                           const float* __restrict__ b2,   // [1]
                           float* __restrict__ out,        // [E]
                           int E)
{
    // W1 in bf16, kept in its native [n][k] order: for the B fragment each
    // lane (= column n) needs two runs of 8 consecutive k values -> two 16B loads.
    __shared__ __bf16 sW1[HIDDEN * W1_LD];                 // ~69.6 KB
    __shared__ __bf16 sPair[WAVES_PB][TILE_M * PAIR_LD];   // ~69.6 KB
    __shared__ float  sB1[HIDDEN];
    __shared__ float  sW2[HIDDEN];

    const int tid      = threadIdx.x;
    const int lane     = tid & 31;
    const int wave     = tid >> 5;
    const int laneHalf = lane >> 4;   // 0: lanes 0-15, 1: lanes 16-31
    const int l16      = lane & 15;

    // ---- stage W1 (fp32 -> bf16), b1, W2 into LDS, block-cooperative ----
    for (int i = tid; i < HIDDEN * KDIM; i += BLOCK_T) {
        int n = i >> 8;           // / KDIM
        int k = i & (KDIM - 1);
        sW1[n * W1_LD + k] = (__bf16)W1[i];
    }
    if (tid < HIDDEN) { sB1[tid] = b1[tid]; sW2[tid] = W2[tid]; }
    __syncthreads();

    // per-lane bias / W2 values for the 8 N-tiles (n = nt*16 + l16)
    float biasReg[8], w2Reg[8];
    #pragma unroll
    for (int nt = 0; nt < 8; ++nt) {
        biasReg[nt] = sB1[nt * 16 + l16];
        w2Reg[nt]   = sW2[nt * 16 + l16];
    }
    const float bias2 = b2[0];

    const int nTiles     = (E + TILE_M - 1) / TILE_M;
    const int waveGlobal = blockIdx.x * WAVES_PB + wave;
    const int waveStride = gridDim.x * WAVES_PB;

    __bf16* myPair = &sPair[wave][0];

    for (int tile = waveGlobal; tile < nTiles; tile += waveStride) {
        // prefetch next tile's edge indices (global_prefetch_b8, speculative)
        __builtin_prefetch(edge_src + (size_t)(tile + waveStride) * TILE_M, 0, 1);
        __builtin_prefetch(edge_dst + (size_t)(tile + waveStride) * TILE_M, 0, 1);

        // ---- gather: lane<16 copies z[src[e]] into cols 0..127 of row l16,
        //              lane>=16 copies z[dst[e]] into cols 128..255 ----
        int e  = tile * TILE_M + l16;
        int ec = e < E ? e : E - 1;
        long long node = laneHalf ? edge_dst[ec] : edge_src[ec];
        const float* zrow = z + (size_t)node * IN_DIM;
        __bf16* dstRow = myPair + l16 * PAIR_LD + laneHalf * IN_DIM;
        #pragma unroll
        for (int c = 0; c < IN_DIM; c += 8) {
            f32x4 a = *(const f32x4*)(zrow + c);
            f32x4 b = *(const f32x4*)(zrow + c + 4);
            bf16x8 h;
            h[0] = (__bf16)a[0]; h[1] = (__bf16)a[1];
            h[2] = (__bf16)a[2]; h[3] = (__bf16)a[3];
            h[4] = (__bf16)b[0]; h[5] = (__bf16)b[1];
            h[6] = (__bf16)b[2]; h[7] = (__bf16)b[3];
            *(bf16x8*)(dstRow + c) = h;
        }
        // LDS ops from one wave execute in order: fragment loads below see
        // the stores above without a barrier (per-wave private region).

        // ---- C[16x128] = A[16x256] x B[256x128] via 8x8 WMMA ----
        f32x8 acc[8];
        #pragma unroll
        for (int nt = 0; nt < 8; ++nt) acc[nt] = (f32x8){};

        #pragma unroll
        for (int kt = 0; kt < KDIM / 32; ++kt) {
            // Opaque zero regenerated every kt: the B-fragment addresses
            // depend on it, so LICM cannot hoist the sW1 LDS loads out of
            // the tile loop (which previously caused 512 VGPRs of
            // loop-invariant B fragments to spill to scratch). The loads
            // stay non-volatile, so the scheduler can still clause and
            // pipeline them against the WMMAs inside one tile iteration.
            int kzero;
            asm volatile("s_mov_b32 %0, 0" : "=s"(kzero));
            const __bf16* w1kt = sW1 + kzero;

            // A fragment (16-bit, 16x32): lanes 0-15 hold K 0-7 & 16-23 of
            // row l16; lanes 16-31 hold K 8-15 & 24-31.
            const __bf16* aBase = myPair + l16 * PAIR_LD + kt * 32 + laneHalf * 8;
            bf16x8 a0 = *(const bf16x8*)(aBase);
            bf16x8 a1 = *(const bf16x8*)(aBase + 16);
            bf16x16 afrag = cat16(a0, a1);

            #pragma unroll
            for (int nt = 0; nt < 8; ++nt) {
                // B fragment (32x16): lane = column n; lanes 0-15 hold
                // K 0-7 & 16-23, lanes 16-31 hold K 8-15 & 24-31.
                const __bf16* bBase =
                    w1kt + (nt * 16 + l16) * W1_LD + kt * 32 + laneHalf * 8;
                bf16x8 b0 = *(const bf16x8*)(bBase);
                bf16x8 b1v = *(const bf16x8*)(bBase + 16);
                bf16x16 bfrag = cat16(b0, b1v);

                acc[nt] = __builtin_amdgcn_wmma_f32_16x16x32_bf16(
                    false, afrag, false, bfrag, (short)0, acc[nt], false, false);
            }
        }

        // ---- epilogue: out[m] = b2 + sum_n relu(H[m][n] + b1[n]) * W2[n]
        // C layout: acc[nt][v] is H[m][n], m = v + laneHalf*8, n = nt*16+l16.
        float res[8];
        #pragma unroll
        for (int v = 0; v < 8; ++v) {
            float s = 0.f;
            #pragma unroll
            for (int nt = 0; nt < 8; ++nt) {
                float x = acc[nt][v] + biasReg[nt];
                x = x > 0.f ? x : 0.f;
                s += x * w2Reg[nt];
            }
            // reduce over the 16 lanes of this half (masks < 16 stay in-half)
            s += __shfl_xor(s, 1, 32);
            s += __shfl_xor(s, 2, 32);
            s += __shfl_xor(s, 4, 32);
            s += __shfl_xor(s, 8, 32);
            res[v] = s;
        }
        #pragma unroll
        for (int v = 0; v < 8; ++v) {
            if (l16 == v) {
                int eo = tile * TILE_M + laneHalf * 8 + v;
                if (eo < E) out[eo] = res[v] + bias2;
            }
        }
    }
}

extern "C" void kernel_launch(void* const* d_in, const int* in_sizes, int n_in,
                              void* d_out, int out_size, void* d_ws, size_t ws_size,
                              hipStream_t stream) {
    const float*     z   = (const float*)d_in[0];
    const long long* ei  = (const long long*)d_in[1];  // int64 [2, E]
    const float*     W1  = (const float*)d_in[2];
    const float*     b1  = (const float*)d_in[3];
    const float*     W2  = (const float*)d_in[4];
    const float*     b2  = (const float*)d_in[5];
    float*           out = (float*)d_out;

    const int E      = in_sizes[1] / 2;          // 600000
    const int nTiles = (E + TILE_M - 1) / TILE_M;
    int blocks = (nTiles + WAVES_PB - 1) / WAVES_PB;
    if (blocks > 1024) blocks = 1024;

    link_pred_wmma_kernel<<<blocks, BLOCK_T, 0, stream>>>(
        z, ei, ei + E, W1, b1, W2, b2, out, E);
}
